// MatrixModel_78134045049038
// MI455X (gfx1250) — matrix operand order repaired and checked by reading, not compiled, not run on hardware
//
#include <hip/hip_runtime.h>
#include <hip/hip_bf16.h>
#include <math.h>
#include <stdint.h>

typedef __attribute__((ext_vector_type(16))) _Float16 v16h;
typedef __attribute__((ext_vector_type(8)))  _Float16 v8h;
typedef __attribute__((ext_vector_type(4)))  _Float16 v4h;
typedef __attribute__((ext_vector_type(8)))  float    v8f;
typedef __attribute__((ext_vector_type(4)))  float    v4f;

#define BM 64
#define BN 256
#define BK 32
#define LDS_PITCH 40   // 32 + 8 halves pad -> conflict-free 16B fragment reads
#define NSLOT 3        // triple-buffered LDS pipeline (single barrier / iter)

__device__ __forceinline__ v16h pack16(v8h lo, v8h hi) {
  v16h r;
  #pragma unroll
  for (int i = 0; i < 8; ++i) { r[i] = lo[i]; r[i + 8] = hi[i]; }
  return r;
}

// C[M,N] = alpha * A[M,K] @ B[N,K]^T (+ R)  -- NT GEMM.
// A: fp32 activations, converted f32->f16 on the LDS-fill path.
// B: f16 (pre-converted weights / f16 activation mirrors), async-copied
//    straight into LDS via global_load_async_to_lds_b128 (ASYNCcnt-tracked),
//    triple-buffered so one full tile of copies is always in flight.
// f16 WMMA, fp32 accumulate. TRANS stores C transposed *as f16* (used for V^T).
template<bool TRANS, bool RES>
__global__ __launch_bounds__(256)
void gemm_nt_wmma(const float* __restrict__ A, const _Float16* __restrict__ B,
                  void* __restrict__ Cv, const float* __restrict__ R,
                  int lda, int ldb, int ldc, int K, float alpha,
                  long long sAb, long long sBb, long long sCb)
{
  __shared__ _Float16 sA[NSLOT][BM][LDS_PITCH];
  __shared__ _Float16 sB[NSLOT][BN][LDS_PITCH];

  const int z = blockIdx.z;
  A += (long long)z * sAb;
  B += (long long)z * sBb;
  float*    C   = nullptr;
  _Float16* C16 = nullptr;
  if constexpr (TRANS) C16 = (_Float16*)Cv + (long long)z * sCb;
  else                 C   = (float*)Cv    + (long long)z * sCb;
  if (RES) R += (long long)z * sCb;

  const int m0 = blockIdx.y * BM;
  const int n0 = blockIdx.x * BN;

  const int t    = threadIdx.x;
  const int lane = t & 31;
  const int wave = t >> 5;
  const int wm   = (wave >> 2) * 32;   // 0 or 32
  const int wn   = (wave & 3) * 64;    // 0,64,128,192
  const int l16  = lane & 15;
  const int hi   = lane >> 4;          // lane-half -> K sub-range per ISA layout

  // cooperative fill coordinates: A 64 rows x 8 floats, B one 32-K row / thread
  const int arow = t >> 2, aseg = (t & 3) * 8;

  const int kIters = K / BK;
  v8f acc[2][4] = {};

  // ---- prologue: stage tile 0 (no readers yet, no barrier needed) ----
  {
    const _Float16* gb = B + (long long)(n0 + t) * ldb;
    #pragma unroll
    for (int c = 0; c < 4; ++c) {
      unsigned ldsoff = (unsigned)(uintptr_t)&sB[0][t][c * 8];
      asm volatile("global_load_async_to_lds_b128 %0, %1, off"
                   :: "v"(ldsoff), "v"(gb + c * 8) : "memory");
    }
    const float* ga = A + (long long)(m0 + arow) * lda + aseg;
    v4f a0 = *(const v4f*)ga;
    v4f a1 = *(const v4f*)(ga + 4);
    v8h ha;
    #pragma unroll
    for (int i = 0; i < 4; ++i) { ha[i] = (_Float16)a0[i]; ha[i + 4] = (_Float16)a1[i]; }
    *(v8h*)&sA[0][arow][aseg] = ha;
  }

  int slot = 0;
  for (int i = 0; i < kIters; ++i) {
    int nslot = slot + 1; if (nslot == NSLOT) nslot = 0;

    if (i + 1 < kIters) {
      const int koff = (i + 1) * BK;
      // async-copy next B tile into the slot whose readers retired 2 barriers ago
      const _Float16* gb = B + (long long)(n0 + t) * ldb + koff;
      #pragma unroll
      for (int c = 0; c < 4; ++c) {
        unsigned ldsoff = (unsigned)(uintptr_t)&sB[nslot][t][c * 8];
        asm volatile("global_load_async_to_lds_b128 %0, %1, off"
                     :: "v"(ldsoff), "v"(gb + c * 8) : "memory");
      }
      __builtin_prefetch(gb + BK, 0, 3);            // global_prefetch_b8 (tile i+2)
      // next A tile: load, convert, ds_store
      const float* ga = A + (long long)(m0 + arow) * lda + koff + aseg;
      v4f a0 = *(const v4f*)ga;
      v4f a1 = *(const v4f*)(ga + 4);
      __builtin_prefetch(ga + BK, 0, 3);
      v8h ha;
      #pragma unroll
      for (int j = 0; j < 4; ++j) { ha[j] = (_Float16)a0[j]; ha[j + 4] = (_Float16)a1[j]; }
      *(v8h*)&sA[nslot][arow][aseg] = ha;
      // tile i's 4 async copies done; tile i+1's 4 stay in flight
      asm volatile("s_wait_asynccnt 0x4" ::: "memory");
    } else {
      asm volatile("s_wait_asynccnt 0x0" ::: "memory");
    }
    __syncthreads();   // single barrier per iteration

    // A fragment: lane half 0 -> K {0..7,16..23}, half 1 -> K {8..15,24..31}
    v16h af[2];
    #pragma unroll
    for (int tm = 0; tm < 2; ++tm) {
      const _Float16* p = &sA[slot][wm + tm * 16 + l16][hi * 8];
      af[tm] = pack16(*(const v8h*)p, *(const v8h*)(p + 16));
    }
    // B fragment: lane half 0 -> K {0..7,16..23}, half 1 -> K {8..15,24..31}
    v16h bf[4];
    #pragma unroll
    for (int tn = 0; tn < 4; ++tn) {
      const _Float16* p = &sB[slot][wn + tn * 16 + l16][hi * 8];
      bf[tn] = pack16(*(const v8h*)p, *(const v8h*)(p + 16));
    }
    #pragma unroll
    for (int tm = 0; tm < 2; ++tm)
      #pragma unroll
      for (int tn = 0; tn < 4; ++tn)
        acc[tm][tn] = __builtin_amdgcn_wmma_f32_16x16x32_f16(
            false, af[tm], false, bf[tn], (short)0, acc[tm][tn], false, false);

    slot = nslot;
  }

  // epilogue: C VGPR j -> row (hi*8 + j), col l16 within each 16x16 tile
  #pragma unroll
  for (int tm = 0; tm < 2; ++tm) {
    #pragma unroll
    for (int tn = 0; tn < 4; ++tn) {
      #pragma unroll
      for (int j = 0; j < 8; ++j) {
        int gm = m0 + wm + tm * 16 + hi * 8 + j;
        int gn = n0 + wn + tn * 16 + l16;
        float v = acc[tm][tn][j] * alpha;
        if constexpr (RES) v += R[(long long)gm * ldc + gn];
        if constexpr (TRANS) C16[(long long)gn * ldc + gm] = (_Float16)v;
        else                 C  [(long long)gm * ldc + gn] = v;
      }
    }
  }
}

__global__ void cvt_weights_f16(const float* __restrict__ src, _Float16* __restrict__ dst) {
  long long i = ((long long)blockIdx.x * blockDim.x + threadIdx.x) * 4;
  v4f x = *(const v4f*)(src + i);
  v4h y;
  #pragma unroll
  for (int j = 0; j < 4; ++j) y[j] = (_Float16)x[j];
  *(v4h*)(dst + i) = y;
}

__global__ void embed_gather(const int* __restrict__ ids, const float* __restrict__ emb,
                             float* __restrict__ h) {
  int row = blockIdx.x;                 // token index in [0, B*S)
  int tok = ids[row];
  const v4f* src = (const v4f*)(emb + (long long)tok * 1024);
  v4f*       dst = (v4f*)(h + (long long)row * 1024);
  dst[threadIdx.x] = src[threadIdx.x];  // 256 threads x float4 = 1024 floats
}

// RoPE. F16OUT=false: rotate x in place (fp32, Q path).
// F16OUT=true: write rotated rows to an f16 mirror (K path -> WMMA B operand).
template<bool F16OUT>
__global__ void rope_rot(float* __restrict__ x, _Float16* __restrict__ y) {
  int idx = blockIdx.x * blockDim.x + threadIdx.x;  // B*S*512 threads
  int d   = idx & 511;
  int row = idx >> 9;
  int s   = row & 1023;
  float freq = __powf(10000.0f, -(float)d * (1.0f / 512.0f));
  float ang  = (float)s * freq;
  float c, sn;
  __sincosf(ang, &sn, &c);
  float* p = x + (long long)row * 1024;
  float x1 = p[d], x2 = p[d + 512];
  float r1 = x1 * c - x2 * sn;
  float r2 = x2 * c + x1 * sn;
  if constexpr (F16OUT) {
    _Float16* py = y + (long long)row * 1024;
    py[d]       = (_Float16)r1;
    py[d + 512] = (_Float16)r2;
  } else {
    p[d]       = r1;
    p[d + 512] = r2;
  }
}

__global__ __launch_bounds__(128)
void softmax_rows(float* __restrict__ x) {   // rows of 1024, one wave per row
  int row  = blockIdx.x * 4 + (threadIdx.x >> 5);
  int lane = threadIdx.x & 31;
  float* p = x + (long long)row * 1024;
  float v[32];
  float m = -3.4e38f;
  #pragma unroll
  for (int i = 0; i < 32; ++i) { v[i] = p[lane + i * 32]; m = fmaxf(m, v[i]); }
  #pragma unroll
  for (int off = 16; off > 0; off >>= 1) m = fmaxf(m, __shfl_xor(m, off, 32));
  float s = 0.0f;
  #pragma unroll
  for (int i = 0; i < 32; ++i) { v[i] = __expf(v[i] - m); s += v[i]; }
  #pragma unroll
  for (int off = 16; off > 0; off >>= 1) s += __shfl_xor(s, off, 32);
  float inv = 1.0f / s;
  #pragma unroll
  for (int i = 0; i < 32; ++i) p[lane + i * 32] = v[i] * inv;
}

__global__ void silu_mul(float* __restrict__ g, const float* __restrict__ u, long long n) {
  long long i = (long long)blockIdx.x * blockDim.x + threadIdx.x;
  if (i < n) {
    float gv = g[i];
    float sg = 1.0f / (1.0f + __expf(-gv));
    g[i] = gv * sg * u[i];
  }
}

extern "C" void kernel_launch(void* const* d_in, const int* in_sizes, int n_in,
                              void* d_out, int out_size, void* d_ws, size_t ws_size,
                              hipStream_t stream) {
  (void)in_sizes; (void)n_in; (void)out_size; (void)ws_size;
  const int* ids      = (const int*)d_in[0];
  const float* blocks = (const float*)d_in[1];

  const int Bb = 2, S = 1024, D = 1024, FF = 4096, Vv = 32000;
  const int M = Bb * S;                       // 2048 token rows

  // Weight offsets mirroring the reference layout (flat fp32 stream).
  const long long DD = (long long)D * D, FD = (long long)FF * D;
  const long long WBASE = (long long)Vv * D;              // weights start after embed
  const long long WCNT  = 2 * (4 * DD + 3 * FD) + WBASE;  // end of layer weights (lm head start)

  // Workspace carve: f16 mirrors first, then fp32 activations.
  _Float16* wh  = (_Float16*)d_ws;            // f16 weights (WCNT-WBASE halves)
  _Float16* kh  = wh + (WCNT - WBASE);        // f16 rotated K  [M][D]
  _Float16* vTh = kh + (long long)M * D;      // f16 V^T        [D][M]
  float* ws = (float*)(vTh + (long long)M * D);
  long long p = 0;
  float* h   = ws + p; p += (long long)M * D;
  float* h2  = ws + p; p += (long long)M * D;
  float* q   = ws + p; p += (long long)M * D;
  float* k   = ws + p; p += (long long)M * D;
  float* ctx = ws + p; p += (long long)M * D;
  float* sc  = ws + p; p += (long long)Bb * S * S;
  float* g   = ws + p; p += (long long)M * FF;
  float* u   = ws + p; p += (long long)M * FF;

  long long off = WBASE;
  const _Float16* W[2][7];
  for (int i = 0; i < 2; ++i) {
    for (int j = 0; j < 4; ++j) { W[i][j] = wh + (off - WBASE); off += DD; }  // q,k,v,o
    for (int j = 4; j < 7; ++j) { W[i][j] = wh + (off - WBASE); off += FD; }  // gate,up,down
  }
  const _Float16* Wlm = wh + (off - WBASE);

  const dim3 blk(256);
  const float scale = 1.0f / 32.0f;            // 1/sqrt(D)

  // One-time (per launch) fp32 -> f16 weight conversion: halves weight bytes
  // and makes every GEMM B-tile a raw async-to-LDS byte copy.
  const long long wn = WCNT - WBASE;           // 33,554,432 elements
  cvt_weights_f16<<<(unsigned)(wn / 1024), 256, 0, stream>>>(blocks + WBASE, wh);

  embed_gather<<<M, 256, 0, stream>>>(ids, blocks, h);

  for (int i = 0; i < 2; ++i) {
    const dim3 gD(D / BN, M / BM, 1);
    // q/k/v projections; V stored transposed + f16 so AV is the same NT pipeline
    gemm_nt_wmma<false,false><<<gD, blk, 0, stream>>>(h, W[i][0], q,   nullptr, D, D, D, D, 1.0f, 0, 0, 0);
    gemm_nt_wmma<false,false><<<gD, blk, 0, stream>>>(h, W[i][1], k,   nullptr, D, D, D, D, 1.0f, 0, 0, 0);
    gemm_nt_wmma<true ,false><<<gD, blk, 0, stream>>>(h, W[i][2], vTh, nullptr, D, D, M, D, 1.0f, 0, 0, 0);
    rope_rot<false><<<(M * (D / 2)) / 256, 256, 0, stream>>>(q, nullptr);  // Q: fp32 in place
    rope_rot<true ><<<(M * (D / 2)) / 256, 256, 0, stream>>>(k, kh);       // K: f16 mirror
    // scores = softmax(q k^T / 32)  (no mask, per reference)
    const dim3 gS(S / BN, S / BM, Bb);
    gemm_nt_wmma<false,false><<<gS, blk, 0, stream>>>(q, kh, sc, nullptr, D, D, S, D, scale,
        (long long)S * D, (long long)S * D, (long long)S * S);
    softmax_rows<<<(Bb * S) / 4, 128, 0, stream>>>(sc);
    // ctx = attn @ V  (NT against f16 V^T; per-batch column window)
    const dim3 gC(D / BN, S / BM, Bb);
    gemm_nt_wmma<false,false><<<gC, blk, 0, stream>>>(sc, vTh, ctx, nullptr, S, M, D, S, 1.0f,
        (long long)S * S, (long long)S, (long long)S * D);
    // h2 = ctx @ Wo^T + h
    gemm_nt_wmma<false,true ><<<gD, blk, 0, stream>>>(ctx, W[i][3], h2, h, D, D, D, D, 1.0f, 0, 0, 0);
    // MLP
    const dim3 gF(FF / BN, M / BM, 1);
    gemm_nt_wmma<false,false><<<gF, blk, 0, stream>>>(h2, W[i][4], g, nullptr, D, D, FF, D, 1.0f, 0, 0, 0);
    gemm_nt_wmma<false,false><<<gF, blk, 0, stream>>>(h2, W[i][5], u, nullptr, D, D, FF, D, 1.0f, 0, 0, 0);
    const long long nsilu = (long long)M * FF;
    silu_mul<<<(unsigned)((nsilu + 255) / 256), 256, 0, stream>>>(g, u, nsilu);
    // h = silu(g)*u @ Wdown^T + h2
    gemm_nt_wmma<false,true ><<<gD, blk, 0, stream>>>(g, W[i][6], h, h2, FF, FF, D, FF, 1.0f, 0, 0, 0);
  }

  // logits = h @ Wlm^T  -> d_out [2,1024,32000]
  const dim3 gL(Vv / BN, M / BM, 1);
  gemm_nt_wmma<false,false><<<gL, blk, 0, stream>>>(h, Wlm, (float*)d_out, nullptr, D, D, Vv, D, 1.0f, 0, 0, 0);
}
